// GRUAgg_34737695490531
// MI455X (gfx1250) — matrix operand order repaired and checked by reading, not compiled, not run on hardware
//
#include <hip/hip_runtime.h>
#include <hip/hip_bf16.h>
#include <math.h>

typedef __attribute__((ext_vector_type(16))) _Float16 v16h;
typedef __attribute__((ext_vector_type(8)))  _Float16 v8h;
typedef __attribute__((ext_vector_type(8)))  float    v8f;

#define AS3 __attribute__((address_space(3)))   // LDS
#define AS1 __attribute__((address_space(1)))   // global

#define BGROUPS 2048
#define LMAX    96
#define DDIM    256
#define G3      768            // 3*DDIM
#define MT      16             // batch tile (WMMA M)
#define NTILES  48             // G3/16
#define KTILES  8              // DDIM/32
#define JPW     6              // N-tiles per wave (48 / 8 waves)

// ---------------------------------------------------------------------------
// Kernel 1: per-group stable sort by timestamp -> permutation [B, LMAX]
// ---------------------------------------------------------------------------
__global__ void build_perm_kernel(const int* __restrict__ index,
                                  const float* __restrict__ ts_in,
                                  int* __restrict__ perm, int E) {
    const int b   = blockIdx.x;
    const int tid = threadIdx.x;
    __shared__ int   s_start, s_count;
    __shared__ float ts[LMAX];

    if (tid == 0) {
        int lo = 0, hi = E;
        while (lo < hi) { int mid = (lo + hi) >> 1; if (index[mid] <  b) lo = mid + 1; else hi = mid; }
        int st = lo;
        lo = st; hi = E;
        while (lo < hi) { int mid = (lo + hi) >> 1; if (index[mid] <= b) lo = mid + 1; else hi = mid; }
        int cnt = lo - st;
        if (cnt > LMAX) cnt = LMAX;
        s_start = st; s_count = cnt;
    }
    __syncthreads();
    const int start = s_start, count = s_count;

    if (tid < count) ts[tid] = ts_in[start + tid];
    __syncthreads();

    for (int s = tid; s < LMAX - count; s += blockDim.x) perm[b * LMAX + s] = -1;
    if (tid < count) {
        float mt = ts[tid];
        int rank = 0;
        for (int k = 0; k < count; ++k) {
            float tk = ts[k];
            rank += (tk < mt) || (tk == mt && k < tid);
        }
        perm[b * LMAX + (LMAX - count) + rank] = start + tid;
    }
}

// ---------------------------------------------------------------------------
// Kernel 2: f32 -> f16 weight conversion
// ---------------------------------------------------------------------------
__global__ void cvt_f32_f16_kernel(const float* __restrict__ src,
                                   _Float16* __restrict__ dst, int n) {
    int i = blockIdx.x * blockDim.x + threadIdx.x;
    if (i < n) dst[i] = (_Float16)src[i];
}

// ---------------------------------------------------------------------------
// WMMA fragment loaders (ISA 7.12.2, wave32)
// ---------------------------------------------------------------------------
__device__ inline v16h frag_a(const AS3 _Float16* rowbase, int hi, int kt) {
    const AS3 _Float16* p = rowbase + kt * 32 + hi * 16;
    v8h lo = *(const AS3 v8h*)p;
    v8h hh = *(const AS3 v8h*)(p + 8);
    v16h r;
#pragma unroll
    for (int i = 0; i < 8; ++i) { r[i] = lo[i]; r[i + 8] = hh[i]; }
    return r;
}

__device__ inline v16h frag_b(const AS1 _Float16* rowbase, int hi, int kt) {
    return *(const AS1 v16h*)(rowbase + kt * 32 + hi * 16);
}

// ---------------------------------------------------------------------------
// One GEMM pass: gout[16,768] = inbuf[16,256] @ W^T + bias, wave handles 6
// N-tiles. kt loop kept rolled so the live set stays ~90 VGPRs (no spills).
// ---------------------------------------------------------------------------
__device__ inline void gemm_pass(const AS3 _Float16* inbuf,
                                 const AS1 _Float16* w,
                                 const AS1 float* bias,
                                 AS3 float* gout) {
    const int tid  = threadIdx.x;
    const int lane = tid & 31;
    const int wave = tid >> 5;
    const int m16  = lane & 15;
    const int hi   = lane >> 4;
    const int n0   = wave * JPW * 16 + m16;

    const AS3 _Float16* arow = inbuf + m16 * DDIM;

    v8f acc[JPW];
#pragma unroll
    for (int jj = 0; jj < JPW; ++jj) {
        float bv = bias[n0 + jj * 16];
#pragma unroll
        for (int i = 0; i < 8; ++i) acc[jj][i] = bv;
    }

#pragma unroll 1
    for (int kt = 0; kt < KTILES; ++kt) {
        v16h a = frag_a(arow, hi, kt);
#pragma unroll
        for (int jj = 0; jj < JPW; ++jj) {
            v16h b = frag_b(w + (size_t)(n0 + jj * 16) * DDIM, hi, kt);
            acc[jj] = __builtin_amdgcn_wmma_f32_16x16x32_f16(false, a, false, b,
                                                             (short)0, acc[jj],
                                                             false, false);
        }
    }

    const int mb = hi * 8;
#pragma unroll
    for (int jj = 0; jj < JPW; ++jj) {
        const int n = n0 + jj * 16;
#pragma unroll
        for (int i = 0; i < 8; ++i) gout[(mb + i) * G3 + n] = acc[jj][i];
    }
}

// ---------------------------------------------------------------------------
// One GRU layer, one time step, for a 16-batch tile.
// ---------------------------------------------------------------------------
template <bool ACCUM>
__device__ inline void gru_layer_step(const AS3 _Float16* inbuf,
                                      AS3 _Float16* hbuf,
                                      AS3 float* gx, AS3 float* gh,
                                      const AS1 _Float16* wih,
                                      const AS1 _Float16* whh,
                                      const AS1 float* bih,
                                      const AS1 float* bhh,
                                      AS3 float* hsum) {
    gemm_pass(inbuf, wih, bih, gx);   // input-gate GEMM
    gemm_pass(hbuf,  whh, bhh, gh);   // recurrent GEMM (disjoint LDS outputs)
    __syncthreads();

    const int tid = threadIdx.x;
    // elementwise gate math: r,z,n at column offsets 0,256,512
#pragma unroll
    for (int k = 0; k < 16; ++k) {
        int e = tid + k * 256;
        int m = e >> 8, c = e & 255;
        float xr = gx[m * G3 + c],            hr = gh[m * G3 + c];
        float xz = gx[m * G3 + DDIM + c],     hz = gh[m * G3 + DDIM + c];
        float xn = gx[m * G3 + 2 * DDIM + c], hn = gh[m * G3 + 2 * DDIM + c];
        float r  = 1.0f / (1.0f + expf(-(xr + hr)));
        float z  = 1.0f / (1.0f + expf(-(xz + hz)));
        float nn = tanhf(xn + r * hn);
        float hp = (float)hbuf[m * DDIM + c];
        float hv = (1.0f - z) * nn + z * hp;
        hbuf[m * DDIM + c] = (_Float16)hv;
        if (ACCUM) hsum[m * DDIM + c] += hv;
    }
    __syncthreads();
}

// ---------------------------------------------------------------------------
// Kernel 3: fused 2-layer GRU + mean pool. One block = 16 batch rows,
// 256 threads = 8 wave32s, persistent over all 96 time steps.
// ---------------------------------------------------------------------------
__global__ __launch_bounds__(256, 1)
void gru_fused_kernel(const float* __restrict__ x_,
                      const int* __restrict__ perm_,
                      const _Float16* __restrict__ wih0_, const _Float16* __restrict__ whh0_,
                      const float* __restrict__ bih0_,    const float* __restrict__ bhh0_,
                      const _Float16* __restrict__ wih1_, const _Float16* __restrict__ whh1_,
                      const float* __restrict__ bih1_,    const float* __restrict__ bhh1_,
                      float* __restrict__ out_) {
    // static LDS: 48+48+16 KB f32, 3x8 KB f16, 64 B = 139328 B (< 320 KB WGP)
    __shared__ float    gx_s[MT * G3];
    __shared__ float    gh_s[MT * G3];
    __shared__ float    hsum_s[MT * DDIM];
    __shared__ _Float16 xbuf_s[MT * DDIM];
    __shared__ _Float16 h1_s[MT * DDIM];
    __shared__ _Float16 h2_s[MT * DDIM];
    __shared__ int      edges_s[MT];

    AS3 float*    gx   = (AS3 float*)gx_s;
    AS3 float*    gh   = (AS3 float*)gh_s;
    AS3 float*    hsum = (AS3 float*)hsum_s;
    AS3 _Float16* xbuf = (AS3 _Float16*)xbuf_s;
    AS3 _Float16* h1   = (AS3 _Float16*)h1_s;
    AS3 _Float16* h2   = (AS3 _Float16*)h2_s;
    AS3 int*      edges = (AS3 int*)edges_s;

    const AS1 float*    x    = (const AS1 float*)x_;
    const AS1 int*      perm = (const AS1 int*)perm_;
    const AS1 _Float16* wih0 = (const AS1 _Float16*)wih0_;
    const AS1 _Float16* whh0 = (const AS1 _Float16*)whh0_;
    const AS1 _Float16* wih1 = (const AS1 _Float16*)wih1_;
    const AS1 _Float16* whh1 = (const AS1 _Float16*)whh1_;
    const AS1 float*    bih0 = (const AS1 float*)bih0_;
    const AS1 float*    bhh0 = (const AS1 float*)bhh0_;
    const AS1 float*    bih1 = (const AS1 float*)bih1_;
    const AS1 float*    bhh1 = (const AS1 float*)bhh1_;
    AS1 float*          out  = (AS1 float*)out_;

    const int tid = threadIdx.x;
    const int bb  = blockIdx.x * MT;

#pragma unroll
    for (int k = 0; k < 16; ++k) {
        int e = tid + k * 256;
        h1[e] = (_Float16)0.0f;
        h2[e] = (_Float16)0.0f;
        hsum[e] = 0.0f;
    }
    __syncthreads();

    for (int t = 0; t < LMAX; ++t) {
        if (tid < MT) edges[tid] = perm[(size_t)(bb + tid) * LMAX + t];
        __syncthreads();
        // gather x rows (f32 -> f16), zeros for pads
#pragma unroll
        for (int k = 0; k < 16; ++k) {
            int e = tid + k * 256;
            int row = e >> 8, col = e & 255;
            int ed = edges[row];
            xbuf[e] = (ed < 0) ? (_Float16)0.0f
                               : (_Float16)x[(size_t)ed * DDIM + col];
        }
        __syncthreads();

        gru_layer_step<false>(xbuf, h1, gx, gh, wih0, whh0, bih0, bhh0, hsum);
        gru_layer_step<true >(h1,   h2, gx, gh, wih1, whh1, bih1, bhh1, hsum);
    }

    const float inv = 1.0f / (float)LMAX;
#pragma unroll
    for (int k = 0; k < 16; ++k) {
        int e = tid + k * 256;
        int row = e >> 8, col = e & 255;
        out[(size_t)(bb + row) * DDIM + col] = hsum[e] * inv;
    }
}

// ---------------------------------------------------------------------------
// Host launcher
// inputs: 0:x 1:timestamps 2:w_ih_l0 3:w_hh_l0 4:b_ih_l0 5:b_hh_l0
//         6:w_ih_l1 7:w_hh_l1 8:b_ih_l1 9:b_hh_l1 10:index
// ---------------------------------------------------------------------------
extern "C" void kernel_launch(void* const* d_in, const int* in_sizes, int n_in,
                              void* d_out, int out_size, void* d_ws, size_t ws_size,
                              hipStream_t stream) {
    const float* x   = (const float*)d_in[0];
    const float* ts  = (const float*)d_in[1];
    const float* w_f32[4] = { (const float*)d_in[2], (const float*)d_in[3],
                              (const float*)d_in[6], (const float*)d_in[7] };
    const float* bih0 = (const float*)d_in[4];
    const float* bhh0 = (const float*)d_in[5];
    const float* bih1 = (const float*)d_in[8];
    const float* bhh1 = (const float*)d_in[9];
    const int*   index = (const int*)d_in[10];
    float* out = (float*)d_out;
    const int E = in_sizes[1];

    char* ws = (char*)d_ws;
    int* perm = (int*)ws;
    size_t off = (size_t)BGROUPS * LMAX * sizeof(int);          // 786432
    _Float16* w_f16[4];
    const int WN = G3 * DDIM;                                   // 196608
    for (int i = 0; i < 4; ++i) {
        w_f16[i] = (_Float16*)(ws + off);
        off += (size_t)WN * sizeof(_Float16);
    }

    build_perm_kernel<<<BGROUPS, 128, 0, stream>>>(index, ts, perm, E);

    for (int i = 0; i < 4; ++i)
        cvt_f32_f16_kernel<<<(WN + 255) / 256, 256, 0, stream>>>(w_f32[i], w_f16[i], WN);

    gru_fused_kernel<<<BGROUPS / MT, 256, 0, stream>>>(
        x, perm,
        w_f16[0], w_f16[1], bih0, bhh0,
        w_f16[2], w_f16[3], bih1, bhh1,
        out);
    (void)n_in; (void)out_size; (void)ws_size;
}
